// VGNN_81097572483251
// MI455X (gfx1250) — matrix operand order repaired and checked by reading, not compiled, not run on hardware
//
#include <hip/hip_runtime.h>
#include <hip/hip_bf16.h>

typedef _Float16 h16;
typedef __attribute__((ext_vector_type(16))) _Float16 v16h;
typedef __attribute__((ext_vector_type(8)))  _Float16 v8h;
typedef __attribute__((ext_vector_type(8)))  float    v8f;

#define PSZ    16
#define IMGSZ  512
#define NPATCH 1024          // (512/16)^2
#define BATCH  16
#define CDIM   256
#define MROWS  (BATCH*NPATCH) // 16384
#define KNN    9

// ---------------------------------------------------------------------------
// WMMA GEMM:  Y[M,N] = X[M,K] @ W[N,K]^T (+bias) (+res1) (+res2)
// X, W are f16; Y is f32 (optional) and f16 (optional).
// One wave computes a 32 x (NT*16) strip (MT=2 rows of 16): every B fragment
// feeds two WMMAs, halving B traffic. NT compile-time -> branch-free K loop
// (WMMA needs EXEC all-ones; all control here is wave-uniform).
// Requires M % 32 == 0 and N % (NT*16) == 0. Batched via blockIdx.z strides.
// ---------------------------------------------------------------------------
template <int NT>
__global__ __launch_bounds__(256)
void gemm_f16_wmma(const h16* __restrict__ Xb, long long sX,
                   const h16* __restrict__ Wb, long long sW,
                   const float* __restrict__ bias,
                   const float* __restrict__ res1,
                   const float* __restrict__ res2,
                   float* __restrict__ Yfb, long long sY,
                   h16* __restrict__ Yh,
                   int M, int N, int K)
{
    const int lane = threadIdx.x & 31;
    const int wid  = (int)((blockIdx.x * blockDim.x + threadIdx.x) >> 5);
    const int tNT  = N / (NT * 16);
    const int totW = (M >> 5) * tNT;          // MT=2 -> 32 rows per wave
    if (wid >= totW) return;                  // wave-uniform -> EXEC stays full
    const int tm  = wid / tNT;
    const int tn  = wid - tm * tNT;

    const h16* X = Xb + (size_t)blockIdx.z * (size_t)sX;
    const h16* W = Wb + (size_t)blockIdx.z * (size_t)sW;

    const int row  = tm * 32 + (lane & 15);   // A-matrix row held by this lane
    const int kbA  = (lane >> 4) * 8;         // A K sub-base (ISA 16-bit A layout)
    const int kbB  = (lane >> 4) * 8;         // B K sub-base (ISA 16-bit layout, same as A)
    const int ncol = (lane & 15);             // B column held by this lane

    const h16* xrow0 = X + (size_t)row * K + kbA;
    const h16* xrow1 = xrow0 + (size_t)16 * K;
    const h16* wbase = W + (size_t)ncol * K + kbB;

    v8f acc[2][NT];
#pragma unroll
    for (int m = 0; m < 2; ++m)
#pragma unroll
        for (int t = 0; t < NT; ++t) acc[m][t] = (v8f){};

    for (int kk = 0; kk < K; kk += 32) {
        v8h a00 = *(const v8h*)(xrow0 + kk);
        v8h a01 = *(const v8h*)(xrow0 + kk + 16);
        v16h A0 = __builtin_shufflevector(a00, a01, 0,1,2,3,4,5,6,7,8,9,10,11,12,13,14,15);
        v8h a10 = *(const v8h*)(xrow1 + kk);
        v8h a11 = *(const v8h*)(xrow1 + kk + 16);
        v16h A1 = __builtin_shufflevector(a10, a11, 0,1,2,3,4,5,6,7,8,9,10,11,12,13,14,15);
#pragma unroll
        for (int t = 0; t < NT; ++t) {
            const h16* wp = wbase + (size_t)((tn * NT + t) * 16) * K + kk;
            v8h b0 = *(const v8h*)(wp);
            v8h b1 = *(const v8h*)(wp + 16);
            v16h Bf = __builtin_shufflevector(b0, b1, 0,1,2,3,4,5,6,7,8,9,10,11,12,13,14,15);
            acc[0][t] = __builtin_amdgcn_wmma_f32_16x16x32_f16(
                false, A0, false, Bf, (short)0, acc[0][t], false, false);
            acc[1][t] = __builtin_amdgcn_wmma_f32_16x16x32_f16(
                false, A1, false, Bf, (short)0, acc[1][t], false, false);
        }
    }

    float* Y = Yfb ? Yfb + (size_t)blockIdx.z * (size_t)sY : nullptr;
    const int rbase = tm * 32 + ((lane >> 4) << 3);   // C/D layout: lanes 16-31 -> M+8
#pragma unroll
    for (int m = 0; m < 2; ++m) {
#pragma unroll
        for (int t = 0; t < NT; ++t) {
            const int col = (tn * NT + t) * 16 + ncol;
            const float bv = bias ? bias[col] : 0.0f;
#pragma unroll
            for (int r = 0; r < 8; ++r) {
                const size_t o = (size_t)(rbase + m * 16 + r) * N + col;
                float v = acc[m][t][r] + bv;
                if (res1) v += res1[o];
                if (res2) v += res2[o];
                if (Y)  Y[o]  = v;
                if (Yh) Yh[o] = (h16)v;
            }
        }
    }
}

// ---------------------------------------------------------------------------
// Patchify: (B,1,512,512) f32 -> (B*N, 256) f16
// ---------------------------------------------------------------------------
__global__ void patchify_k(const float* __restrict__ x, h16* __restrict__ P)
{
    size_t i = (size_t)blockIdx.x * blockDim.x + threadIdx.x;
    if (i >= (size_t)MROWS * CDIM) return;
    int row = (int)(i / CDIM), col = (int)(i % CDIM);
    int b = row / NPATCH, p = row % NPATCH;
    int py = p / 32, px = p % 32, r = col / PSZ, c = col % PSZ;
    P[i] = (h16)x[((size_t)b * IMGSZ + py * PSZ + r) * IMGSZ + px * PSZ + c];
}

__global__ void f32_to_f16_k(const float* __restrict__ s, h16* __restrict__ d, int n)
{
    int i = blockIdx.x * blockDim.x + threadIdx.x;
    if (i < n) d[i] = (h16)s[i];
}

// ---------------------------------------------------------------------------
// BatchNorm (training-mode batch stats, biased var) — column reduction
// ---------------------------------------------------------------------------
__global__ __launch_bounds__(256)
void bn_stats_k(const float* __restrict__ Y, int M, int N,
                float* __restrict__ mean, float* __restrict__ rsig)
{
    __shared__ float ss[256], ss2[256];
    const int c = blockIdx.x;
    float s = 0.f, s2 = 0.f;
    for (int r = threadIdx.x; r < M; r += 256) {
        float v = Y[(size_t)r * N + c];
        s += v; s2 += v * v;
    }
    ss[threadIdx.x] = s; ss2[threadIdx.x] = s2;
    __syncthreads();
    for (int off = 128; off > 0; off >>= 1) {
        if ((int)threadIdx.x < off) { ss[threadIdx.x] += ss[threadIdx.x + off];
                                      ss2[threadIdx.x] += ss2[threadIdx.x + off]; }
        __syncthreads();
    }
    if (threadIdx.x == 0) {
        float m = ss[0] / (float)M;
        float v = ss2[0] / (float)M - m * m;
        mean[c] = m;
        rsig[c] = rsqrtf(v + 1e-5f);
    }
}

__global__ void bn_act_k(const float* __restrict__ Y, const float* __restrict__ mean,
                         const float* __restrict__ rsig, const float* __restrict__ g,
                         const float* __restrict__ bt, int M, int N, int do_gelu,
                         h16* __restrict__ outh, float* __restrict__ outf)
{
    size_t i = (size_t)blockIdx.x * blockDim.x + threadIdx.x;
    if (i >= (size_t)M * N) return;
    int c = (int)(i % N);
    float v = (Y[i] - mean[c]) * rsig[c] * g[c] + bt[c];
    if (do_gelu) v = 0.5f * v * (1.0f + erff(v * 0.70710678118654752f));
    if (outh) outh[i] = (h16)v;
    if (outf) outf[i] = v;
}

__global__ void gelu_k(const float* __restrict__ in, float* __restrict__ outf,
                       h16* __restrict__ outh, size_t n)
{
    size_t i = (size_t)blockIdx.x * blockDim.x + threadIdx.x;
    if (i >= n) return;
    float v = in[i];
    v = 0.5f * v * (1.0f + erff(v * 0.70710678118654752f));
    outf[i] = v;
    outh[i] = (h16)v;
}

__global__ void add_pose_k(const float* __restrict__ h, const float* __restrict__ pose,
                           float* __restrict__ xf, h16* __restrict__ xh)
{
    size_t i = (size_t)blockIdx.x * blockDim.x + threadIdx.x;
    if (i >= (size_t)MROWS * CDIM) return;
    int n = (int)((i / CDIM) % NPATCH), c = (int)(i % CDIM);
    float v = h[i] + pose[(size_t)n * CDIM + c];
    xf[i] = v;
    xh[i] = (h16)v;
}

// ---------------------------------------------------------------------------
// Top-9 (desc, ties -> lowest index like jax.lax.top_k). One thread per row.
// ---------------------------------------------------------------------------
__global__ void topk9_k(const float* __restrict__ sim, int* __restrict__ graph)
{
    int row = blockIdx.x * blockDim.x + threadIdx.x;
    if (row >= MROWS) return;
    const float* s = sim + (size_t)row * NPATCH;
    float vals[KNN]; int idx[KNN];
#pragma unroll
    for (int k = 0; k < KNN; ++k) { vals[k] = -3.4e38f; idx[k] = 0; }
    for (int m = 0; m < NPATCH; ++m) {
        float v = s[m];
        if (v > vals[KNN - 1]) {
            int j = KNN - 1;
            while (j > 0 && v > vals[j - 1]) { vals[j] = vals[j - 1]; idx[j] = idx[j - 1]; --j; }
            vals[j] = v; idx[j] = m;
        }
    }
#pragma unroll
    for (int k = 0; k < KNN; ++k) graph[(size_t)row * KNN + k] = idx[k];
}

// ---------------------------------------------------------------------------
// Gather neighbors, max-aggregate, interleave [x0,a0,x1,a1,...] -> f16 (M x 512)
// ---------------------------------------------------------------------------
__global__ void gather_agg_k(const float* __restrict__ X1, const int* __restrict__ graph,
                             h16* __restrict__ st)
{
    size_t i = (size_t)blockIdx.x * blockDim.x + threadIdx.x;
    if (i >= (size_t)MROWS * CDIM) return;
    int bn = (int)(i / CDIM), c = (int)(i % CDIM);
    int b = bn / NPATCH;
    float xv = X1[i];
    float m = -3.4e38f;
#pragma unroll
    for (int k = 0; k < KNN; ++k) {
        int g = graph[(size_t)bn * KNN + k];
        float nv = X1[((size_t)b * NPATCH + g) * CDIM + c];
        m = fmaxf(m, nv);
    }
    st[(size_t)bn * (2 * CDIM) + 2 * c]     = (h16)xv;
    st[(size_t)bn * (2 * CDIM) + 2 * c + 1] = (h16)(m - xv);
}

// ---------------------------------------------------------------------------
// Host orchestration
// ---------------------------------------------------------------------------
extern "C" void kernel_launch(void* const* d_in, const int* in_sizes, int n_in,
                              void* d_out, int out_size, void* d_ws, size_t ws_size,
                              hipStream_t stream)
{
    (void)in_sizes; (void)n_in; (void)out_size; (void)ws_size;
    auto F = [&](int i) { return (const float*)d_in[i]; };

    // -------- workspace carve-out --------
    char* wsb = (char*)d_ws;
    size_t off = 0;
    auto alloc = [&](size_t bytes) -> char* {
        char* p = wsb + off;
        off = (off + bytes + 255) & ~(size_t)255;
        return p;
    };
    float* simb  = (float*)alloc((size_t)BATCH * NPATCH * NPATCH * 4);
    int*   graph = (int*)  alloc((size_t)MROWS * KNN * 4);
    float* x32A  = (float*)alloc((size_t)MROWS * CDIM * 4);
    float* x32B  = (float*)alloc((size_t)MROWS * CDIM * 4);
    h16*   x16b  = (h16*)  alloc((size_t)MROWS * CDIM * 2);
    float* big32 = (float*)alloc((size_t)MROWS * 1024 * 4);
    h16*   bigh  = (h16*)  alloc((size_t)MROWS * 1024 * 2);
    float* t32   = (float*)alloc((size_t)MROWS * CDIM * 4);
    h16*   t16   = (h16*)  alloc((size_t)MROWS * CDIM * 2);
    float* u32   = (float*)alloc((size_t)MROWS * CDIM * 4);
    h16*   u16   = (h16*)  alloc((size_t)MROWS * CDIM * 2);
    h16*   st16  = (h16*)  alloc((size_t)MROWS * 2 * CDIM * 2);
    float* meanb = (float*)alloc(4096);
    float* rsigb = (float*)alloc(4096);
    h16*   wslot = (h16*)  alloc((size_t)1024 * CDIM * 2);

    auto cvt = [&](const float* s, h16* d, int n) {
        f32_to_f16_k<<<(n + 255) / 256, 256, 0, stream>>>(s, d, n);
    };
    auto gemm = [&](const h16* X, long long sX, const h16* W, long long sW,
                    const float* bias, const float* r1, const float* r2,
                    float* Yf, long long sY, h16* Yh, int M, int N, int K, int nb) {
        if (N % 64 == 0) {
            int waves = (M >> 5) * (N >> 6);
            dim3 g((waves + 7) / 8, 1, nb);
            gemm_f16_wmma<4><<<g, 256, 0, stream>>>(X, sX, W, sW, bias, r1, r2, Yf, sY, Yh, M, N, K);
        } else if (N % 32 == 0) {
            int waves = (M >> 5) * (N >> 5);
            dim3 g((waves + 7) / 8, 1, nb);
            gemm_f16_wmma<2><<<g, 256, 0, stream>>>(X, sX, W, sW, bias, r1, r2, Yf, sY, Yh, M, N, K);
        } else {
            int waves = (M >> 5) * (N >> 4);
            dim3 g((waves + 7) / 8, 1, nb);
            gemm_f16_wmma<1><<<g, 256, 0, stream>>>(X, sX, W, sW, bias, r1, r2, Yf, sY, Yh, M, N, K);
        }
    };
    const int EW = MROWS * CDIM;                     // elementwise count @ width 256
    auto ew = [&](size_t n) { return dim3((unsigned)((n + 255) / 256)); };

    // -------- input indexing --------
    // 0: x; embed stage i: 1+4i..4+4i (w,b,g,bt); 25: pose;
    // block blk base 26+26*blk: in1(+0) out1(+6) in2(+12) out2(+18): w1,b1,g,bt,w2,b2; fc.w(+24) fc.b(+25)
    const float* xin  = F(0);
    const float* pose = F(25);

    // -------- patchify --------
    patchify_k<<<ew((size_t)EW), 256, 0, stream>>>(xin, x16b);

    // -------- embed MLP: 6 x (Linear -> BN -> [GELU]) --------
    const int dims[7] = {256, 128, 64, 32, 64, 128, 256};
    h16* e_in = x16b;
    h16* e_out = u16;
    for (int i = 0; i < 6; ++i) {
        const float* W  = F(1 + 4 * i);
        const float* Bv = F(2 + 4 * i);
        const float* G  = F(3 + 4 * i);
        const float* BT = F(4 + 4 * i);
        int K = dims[i], N = dims[i + 1];
        cvt(W, wslot, N * K);
        gemm(e_in, 0, wslot, 0, Bv, nullptr, nullptr, big32, 0, nullptr, MROWS, N, K, 1);
        bn_stats_k<<<N, 256, 0, stream>>>(big32, MROWS, N, meanb, rsigb);
        if (i < 5) {
            bn_act_k<<<ew((size_t)MROWS * N), 256, 0, stream>>>(
                big32, meanb, rsigb, G, BT, MROWS, N, 1, e_out, nullptr);
            h16* tmp = e_in; e_in = e_out; e_out = tmp;
        } else {
            bn_act_k<<<ew((size_t)MROWS * N), 256, 0, stream>>>(
                big32, meanb, rsigb, G, BT, MROWS, N, 0, nullptr, t32);
        }
    }
    add_pose_k<<<ew((size_t)EW), 256, 0, stream>>>(t32, pose, x32A, x16b);

    // -------- TLN helper: Linear -> BN -> GELU -> Linear (+res1 +res2) --------
    auto tln = [&](const h16* xin16, const float* res1, const float* res2,
                   const float* W1, const float* B1, const float* G, const float* BT,
                   const float* W2, const float* B2, int H,
                   float* outf, h16* outh) {
        cvt(W1, wslot, H * CDIM);
        gemm(xin16, 0, wslot, 0, B1, nullptr, nullptr, big32, 0, nullptr, MROWS, H, CDIM, 1);
        bn_stats_k<<<H, 256, 0, stream>>>(big32, MROWS, H, meanb, rsigb);
        bn_act_k<<<ew((size_t)MROWS * H), 256, 0, stream>>>(
            big32, meanb, rsigb, G, BT, MROWS, H, 1, bigh, nullptr);
        cvt(W2, wslot, CDIM * H);
        gemm(bigh, 0, wslot, 0, B2, res1, res2, outf, 0, outh, MROWS, CDIM, H, 1);
    };

    // -------- 16 ViG blocks --------
    float* xcur = x32A;
    float* xnxt = x32B;
    for (int blk = 0; blk < 16; ++blk) {
        const int bb = 26 + 26 * blk;
        const float* fcw = F(bb + 24);
        const float* fcb = F(bb + 25);
        auto T = [&](int o, int f) { return F(bb + o + f); }; // o: tln offset, f: field

        // 1) sim = x @ x^T per batch (f16 inputs, f32 out)
        gemm(x16b, (long long)NPATCH * CDIM, x16b, (long long)NPATCH * CDIM,
             nullptr, nullptr, nullptr, simb, (long long)NPATCH * NPATCH, nullptr,
             NPATCH, NPATCH, CDIM, BATCH);
        // 2) top-9 neighbors
        topk9_k<<<MROWS / 256, 256, 0, stream>>>(simb, graph);
        // 3) in1 TLN (residual = x) -> x1 in t32/t16
        tln(x16b, xcur, nullptr, T(0,0), T(0,1), T(0,2), T(0,3), T(0,4), T(0,5),
            CDIM, t32, t16);
        // 4) gather + max-aggregate + interleave -> st16 (M x 512)
        gather_agg_k<<<ew((size_t)EW), 256, 0, stream>>>(t32, graph, st16);
        // 5) fc: st @ fc.w^T + b -> u32
        cvt(fcw, wslot, CDIM * 2 * CDIM);
        gemm(st16, 0, wslot, 0, fcb, nullptr, nullptr, u32, 0, nullptr,
             MROWS, CDIM, 2 * CDIM, 1);
        // 6) g = gelu(fc out)
        gelu_k<<<ew((size_t)EW), 256, 0, stream>>>(u32, u32, u16, (size_t)EW);
        // 7) out1 TLN on g, + g (inner residual) + shortcut x -> x2 in t32/t16
        tln(u16, u32, xcur, T(6,0), T(6,1), T(6,2), T(6,3), T(6,4), T(6,5),
            CDIM, t32, t16);
        // 8) in2 TLN (H=1024, residual = x2) -> t in u32/u16
        tln(t16, t32, nullptr, T(12,0), T(12,1), T(12,2), T(12,3), T(12,4), T(12,5),
            1024, u32, u16);
        // 9) g2 = gelu(t)
        gelu_k<<<ew((size_t)EW), 256, 0, stream>>>(u32, u32, u16, (size_t)EW);
        // 10) out2 TLN (H=1024) + g2 + x2 -> next x (last block writes d_out)
        float* dstf = (blk == 15) ? (float*)d_out : xnxt;
        tln(u16, u32, t32, T(18,0), T(18,1), T(18,2), T(18,3), T(18,4), T(18,5),
            1024, dstf, x16b);

        float* tmp = xcur; xcur = xnxt; xnxt = tmp;
    }
}
